// SOAM_2388001816693
// MI455X (gfx1250) — matrix operand order repaired and checked by reading, not compiled, not run on hardware
//
#include <hip/hip_runtime.h>

#define EPS   1e-5f
#define SLOPE 0.2f

#define Bsz  8
#define Cch  64
#define Hdim 256
#define Wdim 256
#define HW   (Hdim*Wdim)     // 65536
#define CHW  (Cch*HW)        // 4194304 per batch
#define NCOL (Cch*Wdim)      // 16384 columns of the flat (256 x 16384) view

typedef __attribute__((ext_vector_type(16))) __bf16         v16bf;
typedef __attribute__((ext_vector_type(8)))  float          v8f;
typedef __attribute__((ext_vector_type(8)))  unsigned short us8;
typedef __attribute__((ext_vector_type(16))) unsigned short us16;

union BF16Frag { us16 u; v16bf b; };

__device__ __forceinline__ unsigned short f2bf(float f) {
  unsigned int u = __float_as_uint(f);
  u += 0x7FFFu + ((u >> 16) & 1u);           // round-to-nearest-even
  return (unsigned short)(u >> 16);
}

__device__ __forceinline__ unsigned int pack2(unsigned short lo, unsigned short hi) {
  return (unsigned int)lo | ((unsigned int)hi << 16);
}

// Build a 16-element bf16 fragment from two 16-byte chunks (b128 loads).
__device__ __forceinline__ v16bf frag2(const unsigned short* p0,
                                       const unsigned short* p1) {
  us8 lo = *(const us8*)p0;
  us8 hi = *(const us8*)p1;
  BF16Frag r;
#pragma unroll
  for (int i = 0; i < 8; ++i) { r.u[i] = lo[i]; r.u[i + 8] = hi[i]; }
  return r.b;
}

// ---------------------------------------------------------------- weights->bf16
__global__ void k_prep_w(const float* __restrict__ wA, const float* __restrict__ wO,
                         unsigned short* __restrict__ wAbf,
                         unsigned short* __restrict__ wObf) {
  int i = blockIdx.x * 256 + threadIdx.x;    // 4096 total
  wAbf[i] = f2bf(wA[i]);
  wObf[i] = f2bf(wO[i]);
}

// ------------------------------------------------- branch 1: C->1 conv+BN+LReLU
// 4 pixels per thread, float4 loads, packed bf16 stores.
__global__ void k_c1(const float* __restrict__ x,  const float* __restrict__ w1,
                     const float* __restrict__ b1, const float* __restrict__ g1,
                     const float* __restrict__ be1,const float* __restrict__ m1,
                     const float* __restrict__ v1,
                     unsigned short* __restrict__ c1bf,
                     unsigned short* __restrict__ c1tbf) {
  int q   = blockIdx.x * 256 + threadIdx.x;  // 0 .. B*HW/4-1
  int b   = q >> 14;                         // / (HW/4)
  int hw4 = (q & (HW / 4 - 1)) * 4;
  const float* xb = x + (size_t)b * CHW + hw4;
  float a0 = 0.f, a1 = 0.f, a2 = 0.f, a3 = 0.f;
#pragma unroll
  for (int c = 0; c < Cch; ++c) {
    float4 xv = *(const float4*)(xb + (size_t)c * HW);
    float  wv = w1[c];
    a0 += xv.x * wv; a1 += xv.y * wv; a2 += xv.z * wv; a3 += xv.w * wv;
  }
  float s  = g1[0] * rsqrtf(v1[0] + EPS);
  float sh = s * (b1[0] - m1[0]) + be1[0];
  float y[4] = { s * a0 + sh, s * a1 + sh, s * a2 + sh, s * a3 + sh };
  ushort4 pk;
  unsigned short bv[4];
#pragma unroll
  for (int j = 0; j < 4; ++j) {
    float t = y[j];
    t = (t >= 0.f) ? t : SLOPE * t;
    bv[j] = f2bf(t);
  }
  pk.x = bv[0]; pk.y = bv[1]; pk.z = bv[2]; pk.w = bv[3];
  *(ushort4*)(c1bf + (size_t)b * HW + hw4) = pk;        // row-major copy
  int h = hw4 >> 8, w = hw4 & 255;                      // 4 consecutive w, same h
#pragma unroll
  for (int j = 0; j < 4; ++j)
    c1tbf[(size_t)b * HW + (w + j) * Hdim + h] = bv[j]; // transposed copy
}

// ------------------------------------------- Gram matrix logits: A @ A^T (bf16)
// One wave per 16x16 output tile; K = 256 in 8 WMMA steps.
__global__ void __launch_bounds__(32)
k_gram(const unsigned short* __restrict__ A, float* __restrict__ out) {
  int bid = blockIdx.x;                       // B * 256 tiles
  int b   = bid >> 8;
  int t   = bid & 255;
  int mt  = t >> 4, nt = t & 15;
  int L   = threadIdx.x, l16 = L & 15, g = L >> 4;
  const unsigned short* Ab = A + (size_t)b * HW;
  int arow = mt * 16 + l16;
  int brow = nt * 16 + l16;                   // B = A^T -> B-frag reads rows of A
  v8f acc = {};
#pragma unroll
  for (int kk = 0; kk < 256; kk += 32) {
    v16bf af = frag2(Ab + arow * 256 + kk + g * 8,
                     Ab + arow * 256 + kk + 16 + g * 8);
    v16bf bf = frag2(Ab + brow * 256 + kk + g * 8,
                     Ab + brow * 256 + kk + 16 + g * 8);
    acc = __builtin_amdgcn_wmma_f32_16x16x32_bf16(false, af, false, bf,
                                                  (short)0, acc, false, false);
  }
  float* ob = out + (size_t)b * HW;
#pragma unroll
  for (int v = 0; v < 8; ++v)
    ob[(mt * 16 + v + 8 * g) * 256 + nt * 16 + l16] = acc[v];
}

// ------------------------------------------------------- row softmax (256 wide)
__global__ void __launch_bounds__(32)
k_softmax(float* __restrict__ s1, float* __restrict__ s2) {
  int bid = blockIdx.x;                       // 2*B*256 rows
  float* base = (bid < Bsz * 256) ? s1 : s2;
  int row     = (bid < Bsz * 256) ? bid : bid - Bsz * 256;
  float* p = base + (size_t)row * 256;
  int L = threadIdx.x;
  float v[8], mx = -3.4e38f;
#pragma unroll
  for (int i = 0; i < 8; ++i) { v[i] = p[i * 32 + L]; mx = fmaxf(mx, v[i]); }
#pragma unroll
  for (int o = 16; o >= 1; o >>= 1) mx = fmaxf(mx, __shfl_xor(mx, o, 32));
  float sum = 0.f;
#pragma unroll
  for (int i = 0; i < 8; ++i) { v[i] = __expf(v[i] - mx); sum += v[i]; }
#pragma unroll
  for (int o = 16; o >= 1; o >>= 1) sum += __shfl_xor(sum, o, 32);
  float inv = 1.f / sum;
#pragma unroll
  for (int i = 0; i < 8; ++i) p[i * 32 + L] = v[i] * inv;
}

// -------------------------------------------------------- Ssum = bf16(S1 + S2)
__global__ void k_ssum(const float* __restrict__ s1, const float* __restrict__ s2,
                       unsigned short* __restrict__ ss) {
  int i4 = (blockIdx.x * 256 + threadIdx.x) * 4;
  float4 a = *(const float4*)(s1 + i4);
  float4 b = *(const float4*)(s2 + i4);
  ushort4 o;
  o.x = f2bf(a.x + b.x); o.y = f2bf(a.y + b.y);
  o.z = f2bf(a.z + b.z); o.w = f2bf(a.w + b.w);
  *(ushort4*)(ss + i4) = o;
}

// ------------------- 1x1 conv (64x64 channel mix) + BN + LReLU via bf16 WMMA.
// Tile: 64 out-channels x 128 pixels per block; 8 waves, 4 accums each.
// Staging packs channel pairs -> 4-byte LDS stores.
__global__ void __launch_bounds__(256)
k_conv(const float* __restrict__ Xf, const unsigned short* __restrict__ Xbf,
       const unsigned short* __restrict__ Wbf,
       const float* __restrict__ bias, const float* __restrict__ gam,
       const float* __restrict__ bet,  const float* __restrict__ mu,
       const float* __restrict__ var,
       float* __restrict__ Yf, unsigned short* __restrict__ Ybf, int addPrev) {
  __shared__ __align__(16) unsigned short Xs[128 * 72]; // [pixel][channel], padded
  int bid = blockIdx.x;
  int b   = bid >> 9;                 // / 512 pixel-tiles
  int p0  = (bid & 511) << 7;         // * 128
  int tid = threadIdx.x;
  if (Xf) {                           // 1024 channel-pair float4 chunks
#pragma unroll
    for (int i = 0; i < 4; ++i) {
      int pc = tid + i * 256;
      int c  = (pc >> 5) * 2;         // channel pair
      int p4 = (pc & 31) * 4;
      const float* xp = Xf + (size_t)b * CHW + (size_t)c * HW + p0 + p4;
      float4 xa = *(const float4*)xp;
      float4 xb = *(const float4*)(xp + HW);
      *(unsigned int*)&Xs[(p4 + 0) * 72 + c] = pack2(f2bf(xa.x), f2bf(xb.x));
      *(unsigned int*)&Xs[(p4 + 1) * 72 + c] = pack2(f2bf(xa.y), f2bf(xb.y));
      *(unsigned int*)&Xs[(p4 + 2) * 72 + c] = pack2(f2bf(xa.z), f2bf(xb.z));
      *(unsigned int*)&Xs[(p4 + 3) * 72 + c] = pack2(f2bf(xa.w), f2bf(xb.w));
    }
  } else {                            // 512 channel-pair us8 chunks
#pragma unroll
    for (int i = 0; i < 2; ++i) {
      int pc = tid + i * 256;
      int c  = (pc >> 4) * 2;
      int p8 = (pc & 15) * 8;
      const unsigned short* xp = Xbf + (size_t)b * CHW + (size_t)c * HW + p0 + p8;
      us8 xa = *(const us8*)xp;
      us8 xb = *(const us8*)(xp + HW);
#pragma unroll
      for (int e = 0; e < 8; ++e)
        *(unsigned int*)&Xs[(p8 + e) * 72 + c] = pack2(xa[e], xb[e]);
    }
  }
  __syncthreads();
  int wave = tid >> 5, L = tid & 31, l16 = L & 15, g = L >> 4;
  int mb = (wave & 3) * 16;           // out-channel sub-tile
  int nb = (wave >> 2) * 64;          // pixel half
  v8f acc[4] = {{}, {}, {}, {}};
#pragma unroll
  for (int kk = 0; kk < 64; kk += 32) {
    v16bf af = frag2(Wbf + (mb + l16) * 64 + kk + g * 8,
                     Wbf + (mb + l16) * 64 + kk + 16 + g * 8);
#pragma unroll
    for (int s = 0; s < 4; ++s) {
      int n = nb + s * 16 + l16;
      v16bf bf = frag2(&Xs[n * 72 + kk + g * 8],
                       &Xs[n * 72 + kk + 16 + g * 8]);
      acc[s] = __builtin_amdgcn_wmma_f32_16x16x32_bf16(false, af, false, bf,
                                                       (short)0, acc[s], false, false);
    }
  }
  float sc[8], sh[8];
#pragma unroll
  for (int v = 0; v < 8; ++v) {       // fold bias+BN: y = sc*acc + sh
    int m = mb + v + 8 * g;
    float s = gam[m] * rsqrtf(var[m] + EPS);
    sc[v] = s;
    sh[v] = s * (bias[m] - mu[m]) + bet[m];
  }
#pragma unroll
  for (int s = 0; s < 4; ++s)
#pragma unroll
    for (int v = 0; v < 8; ++v) {
      int m = mb + v + 8 * g;
      int n = nb + s * 16 + l16;
      float y = sc[v] * acc[s][v] + sh[v];
      y = (y >= 0.f) ? y : SLOPE * y;
      size_t oi = (size_t)b * CHW + (size_t)m * HW + p0 + n;
      if (addPrev) y += Yf[oi];       // out = C2 + C3
      if (Yf)  Yf[oi]  = y;
      if (Ybf) Ybf[oi] = f2bf(y);
    }
}

// ------------- Rsum = (S1+S2) @ view(C2, 256x16384), per batch, bf16 output.
// Block tile: 64 rows x 256 cols; K=256, double-buffered 32-deep LDS tiles,
// b128 staging loads, k-pair-packed b32 transpose stores, full K unroll
// (64 WMMA/wave).
__global__ void __launch_bounds__(256)
k_biggemm(const unsigned short* __restrict__ Ssum,
          const unsigned short* __restrict__ Bmat,
          unsigned short* __restrict__ Rout) {
  __shared__ __align__(16) unsigned short Bs[2][256 * 40]; // [col][k], padded
  int bid  = blockIdx.x;               // B * 4 * 64
  int b    = bid >> 8;
  int r    = bid & 255;
  int mblk = (r >> 6) * 64;
  int n0   = (r & 63) * 256;
  int tid  = threadIdx.x;
  int wave = tid >> 5, L = tid & 31, l16 = L & 15, g = L >> 4;
  int mb = mblk + (wave & 3) * 16;
  int nh = (wave >> 2) * 128;
  const unsigned short* Sb = Ssum + (size_t)b * HW;
  const unsigned short* Bb = Bmat + (size_t)b * CHW + n0;

  // staging: 512 k-pair chunks (16 k-pairs x 32 n-octets), 2 per thread
  int sk[2], sn[2];
#pragma unroll
  for (int i = 0; i < 2; ++i) {
    int pc = tid + i * 256;
    sk[i] = (pc >> 5) * 2;             // even k
    sn[i] = (pc & 31) * 8;
  }
  us8 rga[2], rgb[2];

  auto gload = [&](int kb) {           // 4 x b128 global loads (rows contiguous)
#pragma unroll
    for (int i = 0; i < 2; ++i) {
      const unsigned short* p = Bb + (size_t)(kb * 32 + sk[i]) * NCOL + sn[i];
      rga[i] = *(const us8*)p;
      rgb[i] = *(const us8*)(p + NCOL);
    }
  };
  auto lstore = [&](int buf) {         // transpose on store, packed b32
#pragma unroll
    for (int i = 0; i < 2; ++i)
#pragma unroll
      for (int e = 0; e < 8; ++e)
        *(unsigned int*)&Bs[buf][(sn[i] + e) * 40 + sk[i]] = pack2(rga[i][e], rgb[i][e]);
  };

  v8f acc[8] = {};
  gload(0);
  lstore(0);
#pragma unroll
  for (int kb = 0; kb < 8; ++kb) {
    if (kb < 7) gload(kb + 1);         // overlap next tile's loads with compute
    if (kb < 6)                        // and hint the K+2 tile toward L2
      __builtin_prefetch(Bb + (size_t)((kb + 2) * 32 + sk[0]) * NCOL + sn[0], 0, 1);
    __syncthreads();                   // buf[kb&1] ready; prev reads of other buf done
    if (kb < 7) lstore((kb + 1) & 1);
    const unsigned short* Bsc = Bs[kb & 1];
    v16bf af = frag2(Sb + (mb + l16) * 256 + kb * 32 + g * 8,
                     Sb + (mb + l16) * 256 + kb * 32 + 16 + g * 8);
#pragma unroll
    for (int s = 0; s < 8; ++s) {
      int n = nh + s * 16 + l16;
      v16bf bf = frag2(&Bsc[n * 40 + g * 8], &Bsc[n * 40 + 16 + g * 8]);
      acc[s] = __builtin_amdgcn_wmma_f32_16x16x32_bf16(false, af, false, bf,
                                                       (short)0, acc[s], false, false);
    }
  }
#pragma unroll
  for (int s = 0; s < 8; ++s)
#pragma unroll
    for (int v = 0; v < 8; ++v) {
      int m = mb + v + 8 * g;
      int n = n0 + nh + s * 16 + l16;
      Rout[(size_t)b * CHW + (size_t)m * NCOL + n] = f2bf(acc[s][v]);
    }
}

extern "C" void kernel_launch(void* const* d_in, const int* in_sizes, int n_in,
                              void* d_out, int out_size, void* d_ws, size_t ws_size,
                              hipStream_t stream) {
  (void)in_sizes; (void)n_in; (void)out_size; (void)ws_size;
  const float* x   = (const float*)d_in[0];
  const float* w1  = (const float*)d_in[1];
  const float* b1  = (const float*)d_in[2];
  const float* g1  = (const float*)d_in[3];
  const float* be1 = (const float*)d_in[4];
  const float* m1  = (const float*)d_in[5];
  const float* v1  = (const float*)d_in[6];
  const float* wA  = (const float*)d_in[7];
  const float* bA  = (const float*)d_in[8];
  const float* gA  = (const float*)d_in[9];
  const float* beA = (const float*)d_in[10];
  const float* mA  = (const float*)d_in[11];
  const float* vA  = (const float*)d_in[12];
  const float* wO  = (const float*)d_in[13];
  const float* bO  = (const float*)d_in[14];
  const float* gO  = (const float*)d_in[15];
  const float* beO = (const float*)d_in[16];
  const float* mO  = (const float*)d_in[17];
  const float* vO  = (const float*)d_in[18];

  float* out = (float*)d_out;                       // (B,C,H,W)
  float* S1  = out + (size_t)Bsz * CHW;             // (B,H,H)
  float* S2  = S1 + (size_t)Bsz * HW;               // (B,W,W)

  char* ws = (char*)d_ws;
  unsigned short* c1bf  = (unsigned short*)ws; ws += (size_t)Bsz * HW * 2;
  unsigned short* c1tbf = (unsigned short*)ws; ws += (size_t)Bsz * HW * 2;
  unsigned short* ssbf  = (unsigned short*)ws; ws += (size_t)Bsz * HW * 2;
  unsigned short* wAbf  = (unsigned short*)ws; ws += 8192;
  unsigned short* wObf  = (unsigned short*)ws; ws += 8192;
  unsigned short* c2bf  = (unsigned short*)ws; ws += (size_t)Bsz * CHW * 2;
  unsigned short* rbf   = (unsigned short*)ws;      // Bsz*CHW*2 bytes

  k_prep_w <<<16, 256, 0, stream>>>(wA, wO, wAbf, wObf);
  k_c1     <<<(Bsz * HW) / 1024, 256, 0, stream>>>(x, w1, b1, g1, be1, m1, v1, c1bf, c1tbf);
  k_gram   <<<Bsz * 256, 32, 0, stream>>>(c1bf,  S1);   // logits of C1 @ C1^T
  k_gram   <<<Bsz * 256, 32, 0, stream>>>(c1tbf, S2);   // logits of C1^T @ C1
  k_softmax<<<2 * Bsz * 256, 32, 0, stream>>>(S1, S2);
  k_ssum   <<<(Bsz * HW) / 1024, 256, 0, stream>>>(S1, S2, ssbf);
  // C2 = lrelu(bn(wA @ x)): f32 -> d_out (temp), bf16 -> ws
  k_conv   <<<Bsz * 512, 256, 0, stream>>>(x, nullptr, wAbf, bA, gA, beA, mA, vA,
                                           out, c2bf, 0);
  // R3 + R4 = (S1+S2) @ view(C2)
  k_biggemm<<<Bsz * 256, 256, 0, stream>>>(ssbf, c2bf, rbf);
  // out = C2 + lrelu(bn(wO @ (R3+R4)))
  k_conv   <<<Bsz * 512, 256, 0, stream>>>(nullptr, rbf, wObf, bO, gO, beO, mO, vO,
                                           out, nullptr, 1);
}
